// MoE_44916767982021
// MI455X (gfx1250) — compile-verified, matrix operand order checked by reading, not run on hardware
//
#include <hip/hip_runtime.h>
#include <math.h>

#define TOKENS 4096
#define DMODEL 768
#define DFF    3072
#define NEXP   8

#define BM  64    // block rows (tokens)
#define BN  128   // block cols
#define BKS 64    // K per LDS stage (2 WMMA K-steps)
#define PK  72    // LDS row pitch in f16 (144B rows: 16B aligned, conflict-light)

typedef __attribute__((ext_vector_type(16))) _Float16 v16h;
typedef __attribute__((ext_vector_type(8)))  _Float16 v8h;
typedef __attribute__((ext_vector_type(4)))  _Float16 v4h;
typedef __attribute__((ext_vector_type(8)))  float    v8f;

union V16U { v16h v; v8h h[2]; };

// One-lane WMMA fragment: 8 contiguous f16 at p, 8 at p+16 (K = {8*half+0..7, 16+8*half+0..7}).
__device__ inline v16h frag_ld(const _Float16* p) {
    V16U u;
    u.h[0] = *(const v8h*)(p);
    u.h[1] = *(const v8h*)(p + 16);
    return u.v;
}

__device__ inline v8f wmma16(v16h a, v16h b, v8f c) {
    return __builtin_amdgcn_wmma_f32_16x16x32_f16(
        false, a, false, b, (short)0, c, false, false);
}

// CDNA5 async copy: 16B global -> LDS, tracked by ASYNCcnt.
__device__ inline void async_b128(_Float16* lds_dst, const _Float16* gsrc) {
    unsigned l = (unsigned)(uintptr_t)lds_dst;            // low 32 bits = LDS offset
    unsigned long long g = (unsigned long long)(uintptr_t)gsrc;
    asm volatile("global_load_async_to_lds_b128 %0, %1, off" :: "v"(l), "v"(g) : "memory");
}
__device__ inline void async_wait0() {
    asm volatile("s_wait_asynccnt 0" ::: "memory");
}

// ---------------- x fp32 -> f16 ----------------
__global__ void moe_cvt_x(const float* __restrict__ x, _Float16* __restrict__ xh, int n) {
    int i = blockIdx.x * blockDim.x + threadIdx.x;
    int stride = gridDim.x * blockDim.x;
    for (; i < n; i += stride) xh[i] = (_Float16)x[i];
}

// ---------------- per-expert weight transpose + f32->f16 ----------------
// src: [R][C] fp32 (expert e at e*R*C)  ->  dst: [C][R] f16 (expert e at e*R*C)
__global__ __launch_bounds__(256)
void moe_tr_cvt(const float* __restrict__ src, _Float16* __restrict__ dst, int R, int C) {
    int e = blockIdx.z;
    src += (size_t)e * R * C;
    dst += (size_t)e * R * C;
    int r0 = blockIdx.y * 32, c0 = blockIdx.x * 32;
    __shared__ float t[32][33];
    int tr = threadIdx.x >> 3;
    int tc = (threadIdx.x & 7) * 4;
    float4 v = *(const float4*)(src + (size_t)(r0 + tr) * C + (c0 + tc));
    t[tr][tc + 0] = v.x; t[tr][tc + 1] = v.y;
    t[tr][tc + 2] = v.z; t[tr][tc + 3] = v.w;
    __syncthreads();
    v4h o;
    o[0] = (_Float16)t[tc + 0][tr];
    o[1] = (_Float16)t[tc + 1][tr];
    o[2] = (_Float16)t[tc + 2][tr];
    o[3] = (_Float16)t[tc + 3][tr];
    *(v4h*)(dst + (size_t)(c0 + tr) * R + (r0 + tc)) = o;
}

// ---------------- gating + routing (one wave per token) ----------------
__global__ __launch_bounds__(256)
void moe_gate_route(const float* __restrict__ x, const float* __restrict__ gw,
                    unsigned* __restrict__ cnt, int* __restrict__ idx,
                    float* __restrict__ gwt) {
    int lane = threadIdx.x & 31;
    int wave = threadIdx.x >> 5;
    int t = blockIdx.x * 8 + wave;

    float acc[NEXP];
#pragma unroll
    for (int e = 0; e < NEXP; e++) acc[e] = 0.f;

    const float* xr = x + (size_t)t * DMODEL;
    for (int d = lane; d < DMODEL; d += 32) {
        float xv = xr[d];
        const float* g = gw + d * NEXP;
#pragma unroll
        for (int e = 0; e < NEXP; e++) acc[e] += xv * g[e];
    }
#pragma unroll
    for (int e = 0; e < NEXP; e++) {
#pragma unroll
        for (int off = 16; off >= 1; off >>= 1)
            acc[e] += __shfl_xor(acc[e], off, 32);
    }
    if (lane == 0) {
        float b1 = -1e30f, b2 = -1e30f; int i1 = 0, i2 = 0;
#pragma unroll
        for (int e = 0; e < NEXP; e++) {
            float s = acc[e];
            if (s > b1)      { b2 = b1; i2 = i1; b1 = s; i1 = e; }
            else if (s > b2) { b2 = s;  i2 = e; }
        }
        float e2 = expf(b2 - b1);
        float w1 = 1.f / (1.f + e2);
        float w2 = e2 * w1;
        unsigned s1 = atomicAdd(&cnt[i1], 1u);
        idx[i1 * TOKENS + s1] = t; gwt[i1 * TOKENS + s1] = w1;
        unsigned s2 = atomicAdd(&cnt[i2], 1u);
        idx[i2 * TOKENS + s2] = t; gwt[i2 * TOKENS + s2] = w2;
    }
}

__global__ void moe_prefix8(const unsigned* __restrict__ cnt, unsigned* __restrict__ offs) {
    unsigned s = 0;
    for (int e = 0; e < NEXP; e++) { offs[e] = s; s += cnt[e]; }
}

// ---------------- FC + exact GELU: h = gelu(gather(x) @ w_fc[e]) ----------------
// wt: transposed f16 weights [e][DFF][DMODEL]
__global__ __launch_bounds__(256)
void moe_fc_gelu(const _Float16* __restrict__ xh, const _Float16* __restrict__ wt,
                 const unsigned* __restrict__ cnt, const unsigned* __restrict__ offs,
                 const int* __restrict__ idx, _Float16* __restrict__ h,
                 const _Float16* __restrict__ zpad) {
    int e = blockIdx.z;
    int ncnt = (int)cnt[e];
    int m0 = blockIdx.x * BM;
    if (m0 >= ncnt) return;
    int noff = (int)offs[e];
    int n0 = blockIdx.y * BN;
    const _Float16* W = wt + (size_t)e * DFF * DMODEL;
    const int* ids = idx + e * TOKENS;

    __shared__ __align__(16) _Float16 la[BM * PK];
    __shared__ __align__(16) _Float16 lb[BN * PK];

    int tid = threadIdx.x, lane = tid & 31, wv = tid >> 5;
    int wm = wv & 1, wn = wv >> 1;           // 2 x 4 waves, 32x32 tile each
    int half = lane >> 4, nn = lane & 15;

    // A staging map: 64 rows x 4 chunks of 16 f16 (32B) each
    int ar = tid >> 2, akc = (tid & 3) * 16;
    bool av = (m0 + ar) < ncnt;
    const _Float16* asrc = av ? (xh + (size_t)ids[m0 + ar] * DMODEL + akc) : zpad;
    _Float16* adst = &la[ar * PK + akc];

    // B staging map: 128 rows x 2 chunks of 32 f16 (64B) each
    int bn = tid >> 1, bkc = (tid & 1) * 32;
    const _Float16* bsrc = W + (size_t)(n0 + bn) * DMODEL + bkc;
    _Float16* bdst = &lb[bn * PK + bkc];

    v8f c00 = {}, c01 = {}, c10 = {}, c11 = {};

    for (int k0 = 0; k0 < DMODEL; k0 += BKS) {
        const _Float16* as = av ? (asrc + k0) : zpad;
        async_b128(adst,      as);
        async_b128(adst + 8,  av ? as + 8 : zpad);
        const _Float16* bs = bsrc + k0;
        async_b128(bdst,       bs);
        async_b128(bdst + 8,   bs + 8);
        async_b128(bdst + 16,  bs + 16);
        async_b128(bdst + 24,  bs + 24);
        async_wait0();
        __syncthreads();
#pragma unroll
        for (int ks = 0; ks < BKS; ks += 32) {
            v16h a0 = frag_ld(&la[(wm * 32 + nn) * PK + ks + half * 8]);
            v16h a1 = frag_ld(&la[(wm * 32 + 16 + nn) * PK + ks + half * 8]);
            v16h b0 = frag_ld(&lb[(wn * 32 + nn) * PK + ks + half * 8]);
            v16h b1 = frag_ld(&lb[(wn * 32 + 16 + nn) * PK + ks + half * 8]);
            c00 = wmma16(a0, b0, c00);
            c01 = wmma16(a0, b1, c01);
            c10 = wmma16(a1, b0, c10);
            c11 = wmma16(a1, b1, c11);
        }
        __syncthreads();
    }

    int cb = n0 + wn * 32;
#pragma unroll
    for (int v = 0; v < 8; v++) {
        int sA = m0 + wm * 32 + half * 8 + v;
        int sB = sA + 16;
        if (sA < ncnt) {
            float z0 = c00[v], z1 = c01[v];
            _Float16* hr = h + (size_t)(noff + sA) * DFF + cb;
            hr[nn]      = (_Float16)(0.5f * z0 * (1.f + erff(z0 * 0.70710678118654752f)));
            hr[16 + nn] = (_Float16)(0.5f * z1 * (1.f + erff(z1 * 0.70710678118654752f)));
        }
        if (sB < ncnt) {
            float z0 = c10[v], z1 = c11[v];
            _Float16* hr = h + (size_t)(noff + sB) * DFF + cb;
            hr[nn]      = (_Float16)(0.5f * z0 * (1.f + erff(z0 * 0.70710678118654752f)));
            hr[16 + nn] = (_Float16)(0.5f * z1 * (1.f + erff(z1 * 0.70710678118654752f)));
        }
    }
}

// ---------------- PROJ + weighted scatter: out[tok] += w * (h @ w_proj[e]) ----------------
// wt: transposed f16 weights [e][DMODEL][DFF]; one launch per expert -> no races.
__global__ __launch_bounds__(256)
void moe_proj_comb(const _Float16* __restrict__ h, const _Float16* __restrict__ wt,
                   const unsigned* __restrict__ cnt, const unsigned* __restrict__ offs,
                   const int* __restrict__ idx, const float* __restrict__ gwt,
                   float* __restrict__ out, const _Float16* __restrict__ zpad, int e) {
    int ncnt = (int)cnt[e];
    int m0 = blockIdx.x * BM;
    if (m0 >= ncnt) return;
    int noff = (int)offs[e];
    int n0 = blockIdx.y * BN;
    const _Float16* W = wt + (size_t)e * DFF * DMODEL;

    __shared__ __align__(16) _Float16 la[BM * PK];
    __shared__ __align__(16) _Float16 lb[BN * PK];

    int tid = threadIdx.x, lane = tid & 31, wv = tid >> 5;
    int wm = wv & 1, wn = wv >> 1;
    int half = lane >> 4, nn = lane & 15;

    int ar = tid >> 2, akc = (tid & 3) * 16;
    bool av = (m0 + ar) < ncnt;
    const _Float16* asrc = av ? (h + (size_t)(noff + m0 + ar) * DFF + akc) : zpad;
    _Float16* adst = &la[ar * PK + akc];

    int bn = tid >> 1, bkc = (tid & 1) * 32;
    const _Float16* bsrc = W + (size_t)(n0 + bn) * DFF + bkc;
    _Float16* bdst = &lb[bn * PK + bkc];

    v8f c00 = {}, c01 = {}, c10 = {}, c11 = {};

    for (int k0 = 0; k0 < DFF; k0 += BKS) {
        const _Float16* as = av ? (asrc + k0) : zpad;
        async_b128(adst,      as);
        async_b128(adst + 8,  av ? as + 8 : zpad);
        const _Float16* bs = bsrc + k0;
        async_b128(bdst,       bs);
        async_b128(bdst + 8,   bs + 8);
        async_b128(bdst + 16,  bs + 16);
        async_b128(bdst + 24,  bs + 24);
        async_wait0();
        __syncthreads();
#pragma unroll
        for (int ks = 0; ks < BKS; ks += 32) {
            v16h a0 = frag_ld(&la[(wm * 32 + nn) * PK + ks + half * 8]);
            v16h a1 = frag_ld(&la[(wm * 32 + 16 + nn) * PK + ks + half * 8]);
            v16h b0 = frag_ld(&lb[(wn * 32 + nn) * PK + ks + half * 8]);
            v16h b1 = frag_ld(&lb[(wn * 32 + 16 + nn) * PK + ks + half * 8]);
            c00 = wmma16(a0, b0, c00);
            c01 = wmma16(a0, b1, c01);
            c10 = wmma16(a1, b0, c10);
            c11 = wmma16(a1, b1, c11);
        }
        __syncthreads();
    }

    int cb = n0 + wn * 32;
#pragma unroll
    for (int v = 0; v < 8; v++) {
        int sA = m0 + wm * 32 + half * 8 + v;
        int sB = sA + 16;
        if (sA < ncnt) {
            int tok = idx[e * TOKENS + sA];
            float w = gwt[e * TOKENS + sA];
            float* orow = out + (size_t)tok * DMODEL + cb;
            orow[nn]      += w * c00[v];
            orow[16 + nn] += w * c01[v];
        }
        if (sB < ncnt) {
            int tok = idx[e * TOKENS + sB];
            float w = gwt[e * TOKENS + sB];
            float* orow = out + (size_t)tok * DMODEL + cb;
            orow[nn]      += w * c10[v];
            orow[16 + nn] += w * c11[v];
        }
    }
}

extern "C" void kernel_launch(void* const* d_in, const int* in_sizes, int n_in,
                              void* d_out, int out_size, void* d_ws, size_t ws_size,
                              hipStream_t stream) {
    (void)in_sizes; (void)n_in; (void)out_size; (void)ws_size;
    const float* x      = (const float*)d_in[0];
    const float* gate_w = (const float*)d_in[1];
    const float* w_fc   = (const float*)d_in[2];
    const float* w_proj = (const float*)d_in[3];
    float* out = (float*)d_out;

    // ws layout
    char* ws = (char*)d_ws;
    unsigned* cnt  = (unsigned*)(ws);
    unsigned* offs = (unsigned*)(ws + 32);
    int*      idx  = (int*)(ws + 64);
    float*    gwt  = (float*)(ws + 64 + NEXP * TOKENS * 4);
    size_t o = 64 + (size_t)2 * NEXP * TOKENS * 4;
    o = (o + 255) & ~(size_t)255;
    _Float16* zpad = (_Float16*)(ws + o);                      // 256B zeros
    o += 256;
    _Float16* xh = (_Float16*)(ws + o);                        // 6.3 MB
    o += (size_t)TOKENS * DMODEL * 2;
    o = (o + 255) & ~(size_t)255;
    _Float16* wtfc = (_Float16*)(ws + o);                      // 37.7 MB: [e][DFF][DMODEL]
    o += (size_t)NEXP * DMODEL * DFF * 2;
    _Float16* wtpj = (_Float16*)(ws + o);                      // 37.7 MB: [e][DMODEL][DFF]
    o += (size_t)NEXP * DMODEL * DFF * 2;
    o = (o + 255) & ~(size_t)255;
    _Float16* hbuf = (_Float16*)(ws + o);                      // 50.3 MB: 2*TOKENS x DFF

    hipMemsetAsync(out, 0, (size_t)TOKENS * DMODEL * 4, stream);
    hipMemsetAsync(cnt, 0, NEXP * 4, stream);
    hipMemsetAsync(zpad, 0, 256, stream);

    moe_cvt_x<<<dim3(1536), dim3(256), 0, stream>>>(x, xh, TOKENS * DMODEL);
    // w_fc [D][F] -> wtfc [F][D]
    moe_tr_cvt<<<dim3(DFF / 32, DMODEL / 32, NEXP), dim3(256), 0, stream>>>(w_fc, wtfc, DMODEL, DFF);
    // w_proj [F][D] -> wtpj [D][F]
    moe_tr_cvt<<<dim3(DMODEL / 32, DFF / 32, NEXP), dim3(256), 0, stream>>>(w_proj, wtpj, DFF, DMODEL);
    moe_gate_route<<<dim3(TOKENS / 8), dim3(256), 0, stream>>>(x, gate_w, cnt, idx, gwt);
    moe_prefix8<<<1, 1, 0, stream>>>(cnt, offs);

    moe_fc_gelu<<<dim3(TOKENS / BM, DFF / BN, NEXP), dim3(256), 0, stream>>>(
        xh, wtfc, cnt, offs, idx, hbuf, zpad);
    for (int e = 0; e < NEXP; e++) {
        moe_proj_comb<<<dim3(TOKENS / BM, DMODEL / BN), dim3(256), 0, stream>>>(
            hbuf, wtpj, cnt, offs, idx, gwt, out, zpad, e);
    }
}
